// CausalMultiHeadAttention_89309549953631
// MI455X (gfx1250) — matrix operand order repaired and checked by reading, not compiled, not run on hardware
//
#include <hip/hip_runtime.h>
#include <hip/hip_bf16.h>

typedef __attribute__((ext_vector_type(16))) _Float16 v16h;
typedef __attribute__((ext_vector_type(8)))  _Float16 v8h;
typedef __attribute__((ext_vector_type(8)))  float    v8f;
typedef __attribute__((ext_vector_type(4)))  float    v4f;

namespace {
constexpr int   kD    = 1024;
constexpr int   kH    = 16;
constexpr int   kHD   = 64;
constexpr int   kB    = 2;
constexpr int   kS    = 2048;
constexpr int   kTok  = kB * kS;                     // 4096 tokens
constexpr float kEps  = 1e-5f;
constexpr int   kQT   = kS / 16;                     // 128 query tiles per (b,h)
constexpr int   kKG   = kQT / 4;                     // 32 key groups (64 keys each)
}

// ---------------- WMMA fragment helpers (ISA 7.12.2 layouts, wave32) --------

// A fragment 16x32 f16: lane m = lane&15; lanes>=16 hold K offset +8.
// Per lane: two contiguous 8-halve chunks at k0(+8) and k0+16(+8).
__device__ __forceinline__ v16h frag_a_f16(const _Float16* base, int lda,
                                           int m0, int k0, int lane) {
  int row = m0 + (lane & 15);
  int kb  = k0 + ((lane >> 4) << 3);
  const _Float16* p = base + (size_t)row * lda + kb;
  v8h lo = *(const v8h*)(p);
  v8h hi = *(const v8h*)(p + 16);
  v16h r;
#pragma unroll
  for (int i = 0; i < 8; ++i) { r[i] = lo[i]; r[8 + i] = hi[i]; }
  return r;
}

// Same A fragment but source is fp32 (attn probabilities) -> convert to f16.
__device__ __forceinline__ v16h frag_a_f32(const float* base, int lda,
                                           int m0, int k0, int lane) {
  int row = m0 + (lane & 15);
  int kb  = k0 + ((lane >> 4) << 3);
  const float* p = base + (size_t)row * lda + kb;
  v4f a0 = *(const v4f*)(p);
  v4f a1 = *(const v4f*)(p + 4);
  v4f a2 = *(const v4f*)(p + 16);
  v4f a3 = *(const v4f*)(p + 20);
  v16h r;
#pragma unroll
  for (int i = 0; i < 4; ++i) {
    r[i]      = (_Float16)a0[i];
    r[4 + i]  = (_Float16)a1[i];
    r[8 + i]  = (_Float16)a2[i];
    r[12 + i] = (_Float16)a3[i];
  }
  return r;
}

// B fragment 32x16 f16 where B[k][n] = base[n][k] (row-major "W row" access).
// Per lane: n = lane&15 fixed, two 8-k chunks at k0(+8) and k0+16(+8), matching A.
__device__ __forceinline__ v16h frag_b_rows(const _Float16* base, int ldb,
                                            int n0, int k0, int lane) {
  int n  = n0 + (lane & 15);
  int kb = k0 + ((lane >> 4) << 3);
  const _Float16* p = base + (size_t)n * ldb + kb;
  v8h lo = *(const v8h*)(p);
  v8h hi = *(const v8h*)(p + 16);
  v16h r;
#pragma unroll
  for (int i = 0; i < 8; ++i) { r[i] = lo[i]; r[8 + i] = hi[i]; }
  return r;
}

__device__ __forceinline__ v8f wmma_f16(v16h a, v16h b, v8f c) {
  return __builtin_amdgcn_wmma_f32_16x16x32_f16(false, a, false, b,
                                                (short)0, c, false, false);
}

// ---------------- kernels ---------------------------------------------------

__global__ void cvt_f32_to_f16(const float* __restrict__ src,
                               _Float16* __restrict__ dst, int n4) {
  int i = blockIdx.x * blockDim.x + threadIdx.x;
  if (i >= n4) return;
  v4f v = *(const v4f*)(src + (size_t)i * 4);
  _Float16* d = dst + (size_t)i * 4;
#pragma unroll
  for (int j = 0; j < 4; ++j) d[j] = (_Float16)v[j];
}

// C = A(f16,[M][K]) @ W(f16,[N][K])^T.  One 16x64 tile per wave: the A
// fragment is loaded once per K-step and reused by 4 WMMAs (4x A traffic cut).
// Output f16, optionally transposed store (for Vt).
__global__ void gemm_f16(const _Float16* __restrict__ A,
                         const _Float16* __restrict__ W,
                         _Float16* __restrict__ C,
                         int Mtiles, int Ngroups, int Kdim, int ldOut,
                         int transposed) {
  int wid  = (blockIdx.x * blockDim.x + threadIdx.x) >> 5;
  int lane = threadIdx.x & 31;
  if (wid >= Mtiles * Ngroups) return;
  int m0 = (wid / Ngroups) * 16;
  int n0 = (wid % Ngroups) * 64;
  v8f acc[4] = {};
  for (int k = 0; k < Kdim; k += 32) {
    v16h a = frag_a_f16(A, Kdim, m0, k, lane);
    if (k + 64 < Kdim) {  // speculative prefetch two K-steps ahead
      __builtin_prefetch(A + (size_t)(m0 + (lane & 15)) * Kdim + k + 64, 0, 1);
      __builtin_prefetch(W + (size_t)(n0 + (lane & 15)) * Kdim + k + 64, 0, 1);
    }
#pragma unroll
    for (int j = 0; j < 4; ++j) {
      v16h b = frag_b_rows(W, Kdim, n0 + j * 16, k, lane);
      acc[j] = wmma_f16(a, b, acc[j]);
    }
  }
  int rb = m0 + ((lane >> 4) << 3);
#pragma unroll
  for (int j = 0; j < 4; ++j) {
    int col = n0 + j * 16 + (lane & 15);
#pragma unroll
    for (int r = 0; r < 8; ++r) {
      if (transposed) C[(size_t)col * ldOut + (rb + r)] = (_Float16)acc[j][r];
      else            C[(size_t)(rb + r) * ldOut + col] = (_Float16)acc[j][r];
    }
  }
}

// Output projection: 16x64 tile per wave, fp32 result + bias.
__global__ void gemm_proj(const _Float16* __restrict__ A,
                          const _Float16* __restrict__ W,
                          const float* __restrict__ bias,
                          float* __restrict__ C,
                          int Mtiles, int Ngroups, int Kdim) {
  int wid  = (blockIdx.x * blockDim.x + threadIdx.x) >> 5;
  int lane = threadIdx.x & 31;
  if (wid >= Mtiles * Ngroups) return;
  int m0 = (wid / Ngroups) * 16;
  int n0 = (wid % Ngroups) * 64;
  v8f acc[4] = {};
  for (int k = 0; k < Kdim; k += 32) {
    v16h a = frag_a_f16(A, Kdim, m0, k, lane);
#pragma unroll
    for (int j = 0; j < 4; ++j) {
      v16h b = frag_b_rows(W, Kdim, n0 + j * 16, k, lane);
      acc[j] = wmma_f16(a, b, acc[j]);
    }
  }
  int rb = m0 + ((lane >> 4) << 3);
#pragma unroll
  for (int j = 0; j < 4; ++j) {
    int col = n0 + j * 16 + (lane & 15);
    float bv = bias[col];
#pragma unroll
    for (int r = 0; r < 8; ++r)
      C[(size_t)(rb + r) * kD + col] = acc[j][r] + bv;
  }
}

// Per-(token, head) LayerNorm over 64 dims, in place on f16, fold 'scale'.
__global__ void ln_head(_Float16* __restrict__ t,
                        const float* __restrict__ gamma,
                        const float* __restrict__ beta, float scale) {
  int warp = (blockIdx.x * blockDim.x + threadIdx.x) >> 5;
  int lane = threadIdx.x & 31;
  if (warp >= kTok * kH) return;
  int tok = warp / kH, h = warp % kH;
  _Float16* p = t + (size_t)tok * kD + h * kHD;
  float x0 = (float)p[lane], x1 = (float)p[lane + 32];
  float s = x0 + x1;
#pragma unroll
  for (int o = 16; o > 0; o >>= 1) s += __shfl_xor(s, o, 32);
  float mu = s * (1.0f / 64.0f);
  float d0 = x0 - mu, d1 = x1 - mu;
  float v = d0 * d0 + d1 * d1;
#pragma unroll
  for (int o = 16; o > 0; o >>= 1) v += __shfl_xor(v, o, 32);
  float inv = rsqrtf(v * (1.0f / 64.0f) + kEps);
  p[lane]      = (_Float16)((d0 * inv * gamma[lane]      + beta[lane])      * scale);
  p[lane + 32] = (_Float16)((d1 * inv * gamma[lane + 32] + beta[lane + 32]) * scale);
}

// Raw causal scores (scale pre-folded into q).  One wave owns a 16-query by
// 64-key strip: q fragments loaded once, up to 8 WMMAs across 4 key tiles.
// Above-diagonal tiles skipped wave-uniformly (EXEC stays all-ones for WMMA).
__global__ void scores(const _Float16* __restrict__ qh,
                       const _Float16* __restrict__ kh,
                       float* __restrict__ attn) {
  int wid  = (blockIdx.x * blockDim.x + threadIdx.x) >> 5;
  int lane = threadIdx.x & 31;
  if (wid >= kB * kH * kQT * kKG) return;
  int tg = wid % kKG;
  int ti = (wid / kKG) % kQT;
  int bh = wid / (kKG * kQT);
  if (tg * 4 > ti) return;                 // whole strip above diagonal
  int b = bh / kH, h = bh % kH;
  const _Float16* qb = qh + (size_t)b * kS * kD + h * kHD;
  const _Float16* kb = kh + (size_t)b * kS * kD + h * kHD;
  v16h a0 = frag_a_f16(qb, kD, ti * 16, 0,  lane);
  v16h a1 = frag_a_f16(qb, kD, ti * 16, 32, lane);
  float* base = attn + (size_t)bh * kS * kS;
  int rb = ti * 16 + ((lane >> 4) << 3);
#pragma unroll
  for (int j = 0; j < 4; ++j) {
    int tj = tg * 4 + j;
    if (tj > ti) break;                    // uniform across wave
    v16h b0 = frag_b_rows(kb, kD, tj * 16, 0,  lane);
    v16h b1 = frag_b_rows(kb, kD, tj * 16, 32, lane);
    v8f acc = {};
    acc = wmma_f16(a0, b0, acc);
    acc = wmma_f16(a1, b1, acc);
    int col = tj * 16 + (lane & 15);
#pragma unroll
    for (int r = 0; r < 8; ++r)
      base[(size_t)(rb + r) * kS + col] = acc[r];
  }
}

// Row softmax in place, causal: cols<=i normalized, cols>i set to 0.
__global__ void row_softmax(float* __restrict__ attn) {
  __shared__ float red[256];
  size_t row = blockIdx.x;
  int i   = (int)(row % kS);
  int tid = threadIdx.x;
  float* p = attn + row * (size_t)kS;
  int n = i + 1;
  float m = -1e30f;
  for (int j = tid; j < n; j += 256) m = fmaxf(m, p[j]);
  red[tid] = m; __syncthreads();
  for (int s = 128; s > 0; s >>= 1) {
    if (tid < s) red[tid] = fmaxf(red[tid], red[tid + s]);
    __syncthreads();
  }
  m = red[0]; __syncthreads();
  float sum = 0.0f;
  for (int j = tid; j < n; j += 256) sum += __expf(p[j] - m);
  red[tid] = sum; __syncthreads();
  for (int s = 128; s > 0; s >>= 1) {
    if (tid < s) red[tid] += red[tid + s];
    __syncthreads();
  }
  float inv = 1.0f / red[0];
  for (int j = tid; j < n; j += 256) p[j] = __expf(p[j] - m) * inv;
  for (int j = n + tid; j < kS; j += 256) p[j] = 0.0f;
}

// O = attn @ V via Vt[d][token].  One wave computes a full 16-query x 64-dim
// head tile: the fp32 prob fragment is loaded once per K-step and feeds 4
// WMMAs, cutting the dominant 256MB prob read stream 4x.  K-loop truncated at
// the causal limit (probs beyond the row are exact zeros).
__global__ void attn_v(const float* __restrict__ attn,
                       const _Float16* __restrict__ vt,
                       _Float16* __restrict__ oh) {
  int wid  = (blockIdx.x * blockDim.x + threadIdx.x) >> 5;
  int lane = threadIdx.x & 31;
  if (wid >= kB * kH * kQT) return;
  int qt = wid % kQT;
  int bh = wid / kQT;
  int b = bh / kH, h = bh % kH;
  const float*    ab = attn + (size_t)bh * kS * kS;
  const _Float16* vb = vt + (size_t)(h * kHD) * kTok + b * kS;
  int kmax = (qt + 1) * 16;
  v8f acc[4] = {};
  for (int k = 0; k < kmax; k += 32) {
    v16h a = frag_a_f32(ab, kS, qt * 16, k, lane);
#pragma unroll
    for (int j = 0; j < 4; ++j) {
      v16h bm = frag_b_rows(vb, kTok, j * 16, k, lane);
      acc[j] = wmma_f16(a, bm, acc[j]);
    }
  }
  int rb = b * kS + qt * 16 + ((lane >> 4) << 3);
#pragma unroll
  for (int j = 0; j < 4; ++j) {
    int col = h * kHD + j * 16 + (lane & 15);
#pragma unroll
    for (int r = 0; r < 8; ++r)
      oh[(size_t)(rb + r) * kD + col] = (_Float16)acc[j][r];
  }
}

// ---------------- host launcher ---------------------------------------------

extern "C" void kernel_launch(void* const* d_in, const int* in_sizes, int n_in,
                              void* d_out, int out_size, void* d_ws, size_t ws_size,
                              hipStream_t stream) {
  const float* x  = (const float*)d_in[0];
  const float* Wq = (const float*)d_in[1];
  const float* Wk = (const float*)d_in[2];
  const float* Wv = (const float*)d_in[3];
  const float* Wp = (const float*)d_in[4];
  const float* bp = (const float*)d_in[5];
  const float* gq = (const float*)d_in[6];
  const float* bq = (const float*)d_in[7];
  const float* gk = (const float*)d_in[8];
  const float* bk = (const float*)d_in[9];

  float* out  = (float*)d_out;
  float* attn = out + (size_t)kTok * kD;        // second tuple element

  char* ws = (char*)d_ws;
  const size_t szX = (size_t)kTok * kD * sizeof(_Float16);   // 8 MB
  const size_t szW = (size_t)kD * kD * sizeof(_Float16);     // 2 MB
  _Float16* xh  = (_Float16*)(ws);
  _Float16* wqh = (_Float16*)(ws + szX);
  _Float16* wkh = (_Float16*)(ws + szX + szW);
  _Float16* wvh = (_Float16*)(ws + szX + 2 * szW);
  _Float16* wph = (_Float16*)(ws + szX + 3 * szW);
  _Float16* qh  = (_Float16*)(ws + szX + 4 * szW);
  _Float16* kh  = (_Float16*)(ws + 2 * szX + 4 * szW);
  _Float16* vt  = (_Float16*)(ws + 3 * szX + 4 * szW);       // [kD][kTok]
  _Float16* oh  = (_Float16*)(ws + 4 * szX + 4 * szW);

  // fp32 -> f16 conversions (vectorized 4/thread)
  cvt_f32_to_f16<<<(kTok * kD / 4 + 255) / 256, 256, 0, stream>>>(x,  xh,  kTok * kD / 4);
  cvt_f32_to_f16<<<(kD * kD / 4 + 255) / 256, 256, 0, stream>>>(Wq, wqh, kD * kD / 4);
  cvt_f32_to_f16<<<(kD * kD / 4 + 255) / 256, 256, 0, stream>>>(Wk, wkh, kD * kD / 4);
  cvt_f32_to_f16<<<(kD * kD / 4 + 255) / 256, 256, 0, stream>>>(Wv, wvh, kD * kD / 4);
  cvt_f32_to_f16<<<(kD * kD / 4 + 255) / 256, 256, 0, stream>>>(Wp, wph, kD * kD / 4);

  // QKV projections: 16x64 tiles -> 256*16 = 4096 waves -> 512 blocks
  const int Ngroups    = kD / 64;                      // 16
  const int gemmBlocks = (kTok / 16) * Ngroups / 8;    // 512
  gemm_f16<<<gemmBlocks, 256, 0, stream>>>(xh, wqh, qh, kTok / 16, Ngroups, kD, kD, 0);
  gemm_f16<<<gemmBlocks, 256, 0, stream>>>(xh, wkh, kh, kTok / 16, Ngroups, kD, kD, 0);
  gemm_f16<<<gemmBlocks, 256, 0, stream>>>(xh, wvh, vt, kTok / 16, Ngroups, kD, kTok, 1);

  // Per-head LayerNorm (q gets 1/sqrt(hd) folded in)
  ln_head<<<kTok * kH / 8, 256, 0, stream>>>(qh, gq, bq, 0.125f);
  ln_head<<<kTok * kH / 8, 256, 0, stream>>>(kh, gk, bk, 1.0f);

  // Causal scores: 2*16*128*32 = 131072 wave-strips -> 16384 blocks
  scores<<<kB * kH * kQT * kKG / 8, 256, 0, stream>>>(qh, kh, attn);

  // Row softmax + causal zeroing, in place on attn output
  row_softmax<<<kB * kH * kS, 256, 0, stream>>>(attn);

  // attn @ V -> oh: 2*16*128 = 4096 waves -> 512 blocks
  attn_v<<<kB * kH * kQT / 8, 256, 0, stream>>>(attn, vt, oh);

  // Output projection + bias -> fp32 out
  gemm_proj<<<gemmBlocks, 256, 0, stream>>>(oh, wph, bp, out, kTok / 16, Ngroups, kD);
}
